// PAM_38817914421359
// MI455X (gfx1250) — matrix operand order repaired and checked by reading, not compiled, not run on hardware
//
#include <hip/hip_runtime.h>
#include <hip/hip_bf16.h>

// ---------------------------------------------------------------------------
// PAM (position attention) for B=4, H=W=64, C=512 on gfx1250 (CDNA5).
//   1) cast X f32 -> bf16 (pure BW)
//   2) Q = Xbf@Wf, K = Xbf@Wh (row-major), V = Xbf@Ww (channel-major [b][c][n])
//      -- all on v_wmma_f32_16x16x32_bf16
//   3) fused flash-style attention, 64 keys per step.  Per block: 2 query
//      tiles x 4 channel groups (128 ch/wave).  K/V tiles double-buffered in
//      LDS and prefetched by the Tensor Data Mover (tensor_load_to_lds with
//      s_wait_tensorcnt 2 pipelining); softmax reductions via fused DPP16.
// ---------------------------------------------------------------------------

typedef __attribute__((ext_vector_type(16))) __bf16   v16bf;
typedef __attribute__((ext_vector_type(8)))  __bf16   v8bf;
typedef __attribute__((ext_vector_type(2)))  __bf16   v2bf;
typedef __attribute__((ext_vector_type(8)))  float    v8f;
typedef __attribute__((ext_vector_type(8)))  unsigned v8u;

static __device__ __forceinline__ __bf16 f2bf(float f) {
    unsigned u = __builtin_bit_cast(unsigned, f);
    unsigned r = u + 0x7FFFu + ((u >> 16) & 1u);
    unsigned short h = (unsigned short)(r >> 16);
    return __builtin_bit_cast(__bf16, h);
}

// Packed f32x2 -> bf16x2 (v_cvt_pk_bf16_f32 when available).
#if __has_builtin(__builtin_amdgcn_cvt_pk_bf16_f32)
static __device__ __forceinline__ v2bf pk_bf16(float a, float b) {
    return __builtin_amdgcn_cvt_pk_bf16_f32(a, b);
}
#else
static __device__ __forceinline__ v2bf pk_bf16(float a, float b) {
    v2bf o; o[0] = f2bf(a); o[1] = f2bf(b); return o;
}
#endif

static __device__ __forceinline__ v8f vzero8() {
    v8f z = {0.f,0.f,0.f,0.f,0.f,0.f,0.f,0.f};
    return z;
}

static __device__ __forceinline__ v8f wmma_bf16(v16bf a, v16bf b, v8f c) {
    return __builtin_amdgcn_wmma_f32_16x16x32_bf16(false, a, false, b,
                                                   (short)0, c, false, false);
}

// ---- fused DPP16 butterfly reductions (one VALU op per stage) -------------
static __device__ __forceinline__ float rowmax16(float v) {
    float t;
    asm("v_max_num_f32_dpp %0, %1, %1 quad_perm:[1,0,3,2] row_mask:0xf bank_mask:0xf"
        : "=v"(t) : "v"(v));
    asm("v_max_num_f32_dpp %0, %1, %1 quad_perm:[2,3,0,1] row_mask:0xf bank_mask:0xf"
        : "=v"(v) : "v"(t));
    asm("v_max_num_f32_dpp %0, %1, %1 row_half_mirror row_mask:0xf bank_mask:0xf"
        : "=v"(t) : "v"(v));
    asm("v_max_num_f32_dpp %0, %1, %1 row_mirror row_mask:0xf bank_mask:0xf"
        : "=v"(v) : "v"(t));
    return v;
}
static __device__ __forceinline__ float rowsum16(float v) {
    float t;
    asm("v_add_f32_dpp %0, %1, %1 quad_perm:[1,0,3,2] row_mask:0xf bank_mask:0xf"
        : "=v"(t) : "v"(v));
    asm("v_add_f32_dpp %0, %1, %1 quad_perm:[2,3,0,1] row_mask:0xf bank_mask:0xf"
        : "=v"(v) : "v"(t));
    asm("v_add_f32_dpp %0, %1, %1 row_half_mirror row_mask:0xf bank_mask:0xf"
        : "=v"(t) : "v"(v));
    asm("v_add_f32_dpp %0, %1, %1 row_mirror row_mask:0xf bank_mask:0xf"
        : "=v"(v) : "v"(t));
    return v;
}

// A-matrix (16x32 bf16) dword slot for VGPR v: contiguous k = 16*half + 2v.
static __device__ __forceinline__ int a_dw(int v, int half) {
    return v + 8 * half;
}
static __device__ __forceinline__ v16bf load_a_frag(const unsigned* row32, int half) {
    v8u u;
    #pragma unroll
    for (int v = 0; v < 8; ++v) u[v] = row32[a_dw(v, half)];
    return __builtin_bit_cast(v16bf, u);
}

// ---------------------------------------------------------------------------
// Tensor Data Mover staging (D# per ISA cdna5 ch.8), guarded by __has_builtin.
// ---------------------------------------------------------------------------
#if __has_builtin(__builtin_amdgcn_tensor_load_to_lds) && __has_builtin(__builtin_amdgcn_s_wait_tensorcnt)
#define PAM_USE_TDM 1
typedef __attribute__((ext_vector_type(4))) unsigned tdm_u32x4;
typedef __attribute__((ext_vector_type(4))) int      tdm_i32x4;
typedef __attribute__((ext_vector_type(8))) int      tdm_i32x8;

// 2D tile load, 2-byte elements.  td*/tile* in elements, stride0 in elements.
static __device__ __forceinline__ void tdm_load_2d(unsigned lds_off,
        unsigned long long gaddr, unsigned td0, unsigned td1,
        unsigned tile0, unsigned tile1, unsigned long long stride0)
{
    tdm_u32x4 g0;
    g0[0] = 1u;                                            // count=1, user mode
    g0[1] = lds_off;                                       // lds_addr (bytes)
    g0[2] = (unsigned)(gaddr & 0xffffffffull);             // global_addr lo
    g0[3] = (unsigned)((gaddr >> 32) & 0x01ffffffull)      // global_addr hi
          | 0x80000000u;                                   // type = 2 (image)
    tdm_i32x8 g1;
    g1[0] = (int)(1u << 16);                               // data_size = 2B
    g1[1] = (int)((td0 & 0xffffu) << 16);                  // tensor_dim0 lo16
    g1[2] = (int)(((td0 >> 16) & 0xffffu) | ((td1 & 0xffffu) << 16));
    g1[3] = (int)(((td1 >> 16) & 0xffffu) | ((tile0 & 0xffffu) << 16));
    g1[4] = (int)(tile1 & 0xffffu);                        // tile_dim1, dim2=0
    g1[5] = (int)(stride0 & 0xffffffffull);                // dim0 stride lo32
    g1[6] = (int)((stride0 >> 32) & 0xffffull);            // dim0 stride hi16
    g1[7] = 0;
    tdm_i32x4 z4 = {0, 0, 0, 0};
#if defined(__clang_major__) && (__clang_major__ >= 23)
    tdm_i32x8 z8 = {0, 0, 0, 0, 0, 0, 0, 0};
    __builtin_amdgcn_tensor_load_to_lds(g0, g1, z4, z4, z8, 0);
#else
    __builtin_amdgcn_tensor_load_to_lds(g0, g1, z4, z4, 0);
#endif
}
#else
#define PAM_USE_TDM 0
#endif

// ---------------------------------------------------------------------------
// Kernel 0: X f32 -> bf16.
// ---------------------------------------------------------------------------
__global__ __launch_bounds__(256)
void cast_f32_bf16(const float* __restrict__ in, __bf16* __restrict__ out)
{
    long long i = ((long long)blockIdx.x * 256 + threadIdx.x) * 8;
    float4 f0 = *(const float4*)(in + i);
    float4 f1 = *(const float4*)(in + i + 4);
    v2bf a = pk_bf16(f0.x, f0.y), b = pk_bf16(f0.z, f0.w);
    v2bf c = pk_bf16(f1.x, f1.y), d = pk_bf16(f1.z, f1.w);
    v8bf o;
    o[0] = a[0]; o[1] = a[1]; o[2] = b[0]; o[3] = b[1];
    o[4] = c[0]; o[5] = c[1]; o[6] = d[0]; o[7] = d[1];
    *(v8bf*)(out + i) = o;
}

// ---------------------------------------------------------------------------
// Kernel 1: Y = Xbf[M,K] @ W[K,Nout].  8 waves, 16x64 tile per wave.
// TR=true writes channel-major [b][c][4096] with 16B stores.
// ---------------------------------------------------------------------------
template <bool TR>
__global__ __launch_bounds__(256)
void gemm_xw_bf16(const __bf16* __restrict__ X, const float* __restrict__ W,
                  __bf16* __restrict__ Y, int M, int K, int Nout)
{
    __shared__ __attribute__((aligned(32))) __bf16 wl[64 * 32];   // W tile^T [c][k]

    const int tid  = threadIdx.x;
    const int wave = tid >> 5;
    const int lane = tid & 31;
    const int half = lane >> 4;
    const int n16  = lane & 15;

    const int rowbase = (blockIdx.x * 8 + wave) * 16;
    const int c0      = blockIdx.y * 64;

    v8f acc[4] = {vzero8(), vzero8(), vzero8(), vzero8()};

    for (int kk = 0; kk < K; kk += 32) {
        __syncthreads();
        for (int idx = tid; idx < 32 * 64; idx += 256) {
            int c = idx & 63;
            int k = idx >> 6;
            wl[c * 32 + k] = f2bf(W[(size_t)(kk + k) * Nout + c0 + c]);
        }
        __syncthreads();

        const unsigned* xr =
            (const unsigned*)(X + (size_t)(rowbase + n16) * K + kk);
        v16bf a = load_a_frag(xr, half);

        #pragma unroll
        for (int t = 0; t < 4; ++t) {
            v16bf bfr = *(const v16bf*)&wl[(16 * t + n16) * 32 + half * 16];
            acc[t] = wmma_bf16(a, bfr, acc[t]);
        }
    }

    if (TR) {
        const int bb = rowbase >> 12;
        const int rb = (rowbase & 4095) + 8 * half;
        #pragma unroll
        for (int t = 0; t < 4; ++t) {
            v8bf o;
            #pragma unroll
            for (int r = 0; r < 8; r += 2) {
                v2bf e = pk_bf16(acc[t][r], acc[t][r + 1]);
                o[r] = e[0]; o[r + 1] = e[1];
            }
            int c = c0 + 16 * t + n16;
            *(v8bf*)&Y[((size_t)bb * Nout + c) * 4096 + rb] = o;
        }
    } else {
        #pragma unroll
        for (int t = 0; t < 4; ++t)
            #pragma unroll
            for (int r = 0; r < 8; r += 2) {
                v2bf e = pk_bf16(acc[t][r], acc[t][r + 1]);
                int m = r + 8 * half;
                size_t base = (size_t)(rowbase + m) * Nout + c0 + 16 * t + n16;
                Y[base]        = e[0];
                Y[base + Nout] = e[1];
            }
    }
}

// ---------------------------------------------------------------------------
// Kernel 2: fused attention.  One block per 32-query slab (grid = 4*128).
// 8 waves = 2 query tiles x 4 channel groups; 64 keys per step.
// LDS: 2x(K 8 KB) + 2x(V 64 KB) + P 16 KB = 160 KB  (2 blocks / WGP).
// TDM prefetches tile i+1 while the matrix pipes chew on tile i.
// ---------------------------------------------------------------------------
__global__ __launch_bounds__(256)
void attn_kernel(const __bf16* __restrict__ Q, const __bf16* __restrict__ Km,
                 const __bf16* __restrict__ Vt,  // channel-major [b][c][4096]
                 const float* __restrict__ x, const float* __restrict__ gamma,
                 float* __restrict__ out)
{
    const int N = 4096, C = 512, D = 64;
    const int NIT = N / 64;

    __shared__ __attribute__((aligned(32))) __bf16 k_lds[2][64 * 64];    // 16 KB
    __shared__ __attribute__((aligned(32))) __bf16 vt_lds[2][512 * 64];  // 128 KB
    __shared__ __attribute__((aligned(32))) __bf16 p_lds[8][16 * 64];    // 16 KB

    const int tid  = threadIdx.x;
    const int wave = tid >> 5;
    const int lane = tid & 31;
    const int half = lane >> 4;
    const int n16  = lane & 15;

    const int b     = blockIdx.x >> 7;                    // 128 slabs per batch
    const int qbase = (blockIdx.x & 127) * 32 + (wave >> 2) * 16;
    const int c0    = (wave & 3) * 128;                   // 128 channels / wave
    const size_t rowQ = (size_t)b * N + qbase;
    const __bf16* kbase = Km + (size_t)b * N * D;
    const __bf16* vbase = Vt + (size_t)b * C * N;

    // Q A-fragments (d=64 -> two K=32 fragments), kept in VGPRs.
    v16bf qa[2];
    {
        const unsigned* qrow = (const unsigned*)(Q + (rowQ + n16) * D);
        qa[0] = load_a_frag(qrow, half);
        qa[1] = load_a_frag(qrow + 16, half);
    }

    v8f oacc[8];
    #pragma unroll
    for (int t = 0; t < 8; ++t) oacc[t] = vzero8();
    float mrun[8], lrun[8];
    #pragma unroll
    for (int r = 0; r < 8; ++r) { mrun[r] = -1e30f; lrun[r] = 0.f; }

#if PAM_USE_TDM
    if (wave == 0) {                       // kick off tile 0 (async)
        tdm_load_2d((unsigned)(unsigned long long)(void*)&k_lds[0][0],
                    (unsigned long long)(const void*)kbase,
                    4096u, 1u, 4096u, 1u, 4096u);
        tdm_load_2d((unsigned)(unsigned long long)(void*)&vt_lds[0][0],
                    (unsigned long long)(const void*)vbase,
                    (unsigned)N, (unsigned)C, 64u, (unsigned)C,
                    (unsigned long long)N);
    }
#endif

    for (int it = 0; it < NIT; ++it) {
        const int j0  = it * 64;
        const int buf = it & 1;
        __syncthreads();   // everyone finished reading buf^1 (iter it-1)
#if PAM_USE_TDM
        if (wave == 0) {
            if (it + 1 < NIT) {
                // prefetch next tile into the other buffer, then wait until
                // only those 2 ops are outstanding (current tile complete).
                tdm_load_2d((unsigned)(unsigned long long)(void*)&k_lds[buf ^ 1][0],
                            (unsigned long long)(const void*)(kbase + (size_t)(j0 + 64) * D),
                            4096u, 1u, 4096u, 1u, 4096u);
                tdm_load_2d((unsigned)(unsigned long long)(void*)&vt_lds[buf ^ 1][0],
                            (unsigned long long)(const void*)(vbase + j0 + 64),
                            (unsigned)N, (unsigned)C, 64u, (unsigned)C,
                            (unsigned long long)N);
                __builtin_amdgcn_s_wait_tensorcnt(2);
            } else {
                __builtin_amdgcn_s_wait_tensorcnt(0);
            }
        }
#else
        {
            const uint4* ks = (const uint4*)(kbase + (size_t)j0 * D);
            ((uint4*)&k_lds[buf][0])[tid]       = ks[tid];
            ((uint4*)&k_lds[buf][0])[tid + 256] = ks[tid + 256];
            const __bf16* vb = vbase + j0;
            for (int idx = tid; idx < 512 * 8; idx += 256) {
                int c = idx >> 3, p = idx & 7;
                ((uint4*)&vt_lds[buf][0])[c * 8 + p] =
                    *(const uint4*)(vb + (size_t)c * N + p * 8);
            }
        }
#endif
        __syncthreads();

        // S = Q (16x64) @ K^T (64x64) -> four 16x16 f32 tiles.
        v8f s[4];
        #pragma unroll
        for (int jt = 0; jt < 4; ++jt) {
            const __bf16* kr = &k_lds[buf][(16 * jt + n16) * 64 + half * 16];
            v8f a = wmma_bf16(qa[0], *(const v16bf*)kr, vzero8());
            s[jt]  = wmma_bf16(qa[1], *(const v16bf*)(kr + 32), a);
        }

        // Online softmax (row m = r + 8*half lives in one 16-lane group).
        __bf16* pw = &p_lds[wave][0];
        #pragma unroll
        for (int r = 0; r < 8; ++r) {
            float v0 = s[0][r], v1 = s[1][r], v2 = s[2][r], v3 = s[3][r];
            float mx   = rowmax16(fmaxf(fmaxf(v0, v1), fmaxf(v2, v3)));
            float mnew = fmaxf(mrun[r], mx);
            float sc   = __expf(mrun[r] - mnew);
            float p0 = __expf(v0 - mnew), p1 = __expf(v1 - mnew);
            float p2 = __expf(v2 - mnew), p3 = __expf(v3 - mnew);
            float rs = rowsum16((p0 + p1) + (p2 + p3));
            lrun[r] = lrun[r] * sc + rs;
            mrun[r] = mnew;
            #pragma unroll
            for (int t = 0; t < 8; ++t) oacc[t][r] *= sc;
            int m = r + 8 * half;
            v2bf q01 = pk_bf16(p0, p1), q23 = pk_bf16(p2, p3);
            pw[m * 64 + n16]      = q01[0];
            pw[m * 64 + 16 + n16] = q01[1];
            pw[m * 64 + 32 + n16] = q23[0];
            pw[m * 64 + 48 + n16] = q23[1];
        }
        __syncthreads();   // C-layout -> A-layout transpose through LDS

        const unsigned* prow = (const unsigned*)&pw[n16 * 64];
        v16bf pa0 = load_a_frag(prow, half);
        v16bf pa1 = load_a_frag(prow + 16, half);

        // O += P (16x64) @ V (64x128): sixteen 16-channel WMMAs.
        #pragma unroll
        for (int t = 0; t < 8; ++t) {
            const __bf16* vc = &vt_lds[buf][(c0 + 16 * t + n16) * 64 + half * 16];
            oacc[t] = wmma_bf16(pa0, *(const v16bf*)vc, oacc[t]);
            oacc[t] = wmma_bf16(pa1, *(const v16bf*)(vc + 32), oacc[t]);
        }
    }

    // Epilogue: out = gamma * (O / l) + x
    const float g = gamma[0];
    #pragma unroll
    for (int t = 0; t < 8; ++t)
        #pragma unroll
        for (int r = 0; r < 8; ++r) {
            int m   = r + 8 * half;
            int col = c0 + 16 * t + n16;
            size_t idx = ((size_t)(b * N + qbase + m)) * C + col;
            out[idx] = g * (oacc[t][r] / lrun[r]) + x[idx];
        }
}

// ---------------------------------------------------------------------------
// Launch.  d_ws layout: Xbf 16 MB | Q 2 MB | K 2 MB | Vt 16 MB  (36 MB total)
// ---------------------------------------------------------------------------
extern "C" void kernel_launch(void* const* d_in, const int* in_sizes, int n_in,
                              void* d_out, int out_size, void* d_ws, size_t ws_size,
                              hipStream_t stream) {
    const float* x     = (const float*)d_in[0];
    const float* Wf    = (const float*)d_in[1];
    const float* Wh    = (const float*)d_in[2];
    const float* Ww    = (const float*)d_in[3];
    const float* gamma = (const float*)d_in[4];
    float* out = (float*)d_out;

    const int M = 4 * 64 * 64;   // 16384 tokens
    const int C = 512, D = 64;

    __bf16* Xbf = (__bf16*)d_ws;
    __bf16* Q   = Xbf + (size_t)M * C;
    __bf16* Kq  = Q   + (size_t)M * D;
    __bf16* Vt  = Kq  + (size_t)M * D;   // [b][c][4096]

    dim3 blk(256);
    cast_f32_bf16<<<dim3((M * C) / (256 * 8)), blk, 0, stream>>>(x, Xbf);

    gemm_xw_bf16<false><<<dim3(M / 128, 1), blk, 0, stream>>>(Xbf, Wf, Q,  M, C, D);
    gemm_xw_bf16<false><<<dim3(M / 128, 1), blk, 0, stream>>>(Xbf, Wh, Kq, M, C, D);
    gemm_xw_bf16<true ><<<dim3(M / 128, C / 64), blk, 0, stream>>>(Xbf, Ww, Vt, M, C, C);

    attn_kernel<<<dim3(4 * 128), blk, 0, stream>>>(Q, Kq, Vt, x, gamma, out);
}
